// MultiHeadAttentionImproved_52364241272887
// MI455X (gfx1250) — matrix operand order repaired and checked by reading, not compiled, not run on hardware
//
#include <hip/hip_runtime.h>
#include <hip/hip_bf16.h>

// ---------------------------------------------------------------------------
// MHA with relative position bias for gfx1250 (MI455X), wave32 + WMMA bf16
// + Tensor Data Mover (TDM) double-buffered V-tile staging.
// B=4, S=1024, D=1024, H=16, dk=64, MAX_REL=32.
// ---------------------------------------------------------------------------

#define B_SZ   4
#define S_LEN  1024
#define D_MOD  1024
#define H_NUM  16
#define HPB    8           // heads per attention block (grid.y = 2)
#define DK     64
#define NREL   65          // 2*MAX_REL+1
#define QT     32          // query rows per attention block

typedef __attribute__((ext_vector_type(16))) __bf16 v16bf;
typedef __attribute__((ext_vector_type(8)))  __bf16 v8bf;
typedef __attribute__((ext_vector_type(8)))  float  v8f;
typedef __attribute__((ext_vector_type(4)))  unsigned int v4u;
typedef __attribute__((ext_vector_type(8)))  int v8i;
typedef __attribute__((ext_vector_type(4)))  int v4i;

#if defined(__has_builtin)
#if __has_builtin(__builtin_amdgcn_tensor_load_to_lds) && __has_builtin(__builtin_amdgcn_s_wait_tensorcnt)
#define USE_TDM 1
#endif
#endif
#ifndef USE_TDM
#define USE_TDM 0
#endif

// D = A(16x32 bf16) * B(32x16 bf16) + C(16x16 f32), wave32.
__device__ __forceinline__ v8f wmma_bf16(v16bf a, v16bf b, v8f c) {
    return __builtin_amdgcn_wmma_f32_16x16x32_bf16(
        /*neg_a=*/false, a, /*neg_b=*/false, b,
        /*c_mod=*/(short)0, c, /*reuse_a=*/false, /*reuse_b=*/false);
}

// A-matrix (16x32, 16-bit) lane layout, ISA 7.12.2:
//   lane half = lane>>4; row M = lane&15
//   element e=0..7  -> K = half*8 + e        (contiguous 8)
//   element e=8..15 -> K = 16 + half*8 + e-8 (contiguous 8)
__device__ __forceinline__ v16bf load_a_frag(const __bf16* rowK0, int lhalf) {
    v8bf lo = *reinterpret_cast<const v8bf*>(rowK0 + lhalf * 8);
    v8bf hi = *reinterpret_cast<const v8bf*>(rowK0 + 16 + lhalf * 8);
    v16bf a;
#pragma unroll
    for (int e = 0; e < 8; ++e) { a[e] = lo[e]; a[e + 8] = hi[e]; }
    return a;
}

#if USE_TDM
// Issue a TDM 2D tile load: rows x cols bf16 tile, global row stride (elems),
// into LDS at ldsAddr. D# per CDNA5 ISA ch.8 (group0 + group1; groups 2/3 and
// the second descriptor slot are zero). 6-arg builtin form (clang-23 lane).
__device__ __forceinline__ void tdm_load_tile_bf16(const __bf16* gsrc,
                                                   unsigned ldsAddr,
                                                   int cols, int rows,
                                                   int rowStride) {
    unsigned long long ga = (unsigned long long)(uintptr_t)gsrc;
    v4u g0;
    g0[0] = 1u;                                   // count=1, user mode
    g0[1] = ldsAddr;                              // lds_addr (bytes)
    g0[2] = (unsigned)(ga & 0xFFFFFFFFu);         // global_addr[31:0]
    g0[3] = (unsigned)((ga >> 32) & 0x1FFFFFFu)   // global_addr[56:32]
            | (2u << 30);                         // type=2 (image)
    v8i g1;
    g1[0] = (1 << 16);                            // data_size=1 (2 bytes)
    g1[1] = (cols & 0xFFFF) << 16;                // tensor_dim0[15:0] @bits63:48
    g1[2] = (rows & 0xFFFF) << 16;                // tensor_dim1[15:0] @bits111:80 (low 16)
    g1[3] = (cols & 0xFFFF) << 16;                // tile_dim0 @bits127:112
    g1[4] = (rows & 0xFFFF);                      // tile_dim1 @bits143:128
    g1[5] = rowStride;                            // tensor_dim0_stride[31:0]
    g1[6] = 0;                                    // stride[47:32]=0, dim1_stride lo=0
    g1[7] = 0;
    v4i z4 = {0, 0, 0, 0};
    v8i z8 = {0, 0, 0, 0, 0, 0, 0, 0};
    __builtin_amdgcn_tensor_load_to_lds(g0, g1, z4, z4, z8, 0);
}
#endif

// ---------------------------------------------------------------------------
__global__ void k_cast_bf16(const float* __restrict__ src,
                            __bf16* __restrict__ dst, size_t n) {
    size_t i = (size_t)blockIdx.x * blockDim.x + threadIdx.x;
    if (i < n) dst[i] = (__bf16)src[i];
}

__global__ void k_zero_f32(float* __restrict__ p, size_t n) {
    size_t i = (size_t)blockIdx.x * blockDim.x + threadIdx.x;
    if (i < n) p[i] = 0.0f;
}

__global__ void k_mean_combine(const float* __restrict__ a,
                               const float* __restrict__ b,
                               float* __restrict__ out, size_t n) {
    size_t i = (size_t)blockIdx.x * blockDim.x + threadIdx.x;
    if (i < n) out[i] = a[i] + b[i];
}

// ---------------------------------------------------------------------------
// C[M,N] = A[M,K] @ Bm[K,N] (+bias), bf16 inputs, f32 accumulate via WMMA.
// 64x64 C tile per block, 8 waves, each wave owns 2 M-subtiles sharing one
// B fragment. Software-pipelined: next-iteration fragments are loaded before
// the current WMMAs so loads overlap matrix math.
__global__ __launch_bounds__(256)
void k_gemm_bf16(const __bf16* __restrict__ A, const __bf16* __restrict__ Bm,
                 const float* __restrict__ bias,
                 float* __restrict__ outF, __bf16* __restrict__ outB,
                 int M, int N, int K) {
    const int m0 = blockIdx.x * 64, n0 = blockIdx.y * 64;
    const int w = threadIdx.x >> 5, lane = threadIdx.x & 31;
    const int lhalf = lane >> 4, lm = lane & 15;

    const int nt  = w & 3;
    const int mt0 = (w >> 2) * 2;
    const int mt1 = mt0 + 1;

    const __bf16* arow0 = A + (size_t)(m0 + mt0 * 16 + lm) * K;
    const __bf16* arow1 = A + (size_t)(m0 + mt1 * 16 + lm) * K;
    const __bf16* bcol  = Bm + n0 + nt * 16;

    v8f acc0 = {}, acc1 = {};
    // prologue loads
    v16bf bf;
#pragma unroll
    for (int e = 0; e < 16; ++e) { int kk = (e < 8) ? (lhalf * 8 + e) : (16 + lhalf * 8 + (e - 8)); bf[e] = bcol[(size_t)kk * N + lm]; }
    v16bf a0 = load_a_frag(arow0, lhalf);
    v16bf a1 = load_a_frag(arow1, lhalf);
    for (int k0 = 0; k0 < K; k0 += 32) {
        const int kn = (k0 + 32 < K) ? k0 + 32 : k0;     // clamp on epilogue
        const int kp = (kn + 32 < K) ? kn + 32 : kn;
        v16bf bfn;
#pragma unroll
        for (int e = 0; e < 16; ++e) { int kk = (e < 8) ? (lhalf * 8 + e) : (16 + lhalf * 8 + (e - 8)); bfn[e] = bcol[(size_t)(kn + kk) * N + lm]; }
        v16bf a0n = load_a_frag(arow0 + kn, lhalf);
        v16bf a1n = load_a_frag(arow1 + kn, lhalf);
        __builtin_prefetch(bcol + (size_t)(kp + lane) * N, 0, 1);
        acc0 = wmma_bf16(a0, bf, acc0);
        acc1 = wmma_bf16(a1, bf, acc1);
        bf = bfn; a0 = a0n; a1 = a1n;
    }
    const int col = n0 + nt * 16 + lm;
    const float bv = bias ? bias[col] : 0.0f;
#pragma unroll
    for (int v = 0; v < 8; ++v) {
        int r0 = m0 + mt0 * 16 + v + 8 * lhalf;
        int r1 = m0 + mt1 * 16 + v + 8 * lhalf;
        float v0 = acc0[v] + bv, v1 = acc1[v] + bv;
        if (outF) { outF[(size_t)r0 * N + col] = v0; outF[(size_t)r1 * N + col] = v1; }
        if (outB) { outB[(size_t)r0 * N + col] = (__bf16)v0; outB[(size_t)r1 * N + col] = (__bf16)v1; }
    }
}

// ---------------------------------------------------------------------------
// QR[b,h,i,r] = sum_d Qf[b,i,h*64+d] * rel_k[r,d]
__global__ void k_qrel(const float* __restrict__ Qf,
                       const float* __restrict__ relk,
                       float* __restrict__ QR) {
    size_t gid = (size_t)blockIdx.x * blockDim.x + threadIdx.x;
    const size_t total = (size_t)B_SZ * H_NUM * S_LEN * NREL;
    if (gid >= total) return;
    int r  = (int)(gid % NREL);
    size_t t = gid / NREL;
    int i  = (int)(t % S_LEN);
    int bh = (int)(t / S_LEN);
    int b = bh >> 4, h = bh & 15;
    const float* q  = Qf + ((size_t)(b * S_LEN + i)) * D_MOD + h * DK;
    const float* rk = relk + r * DK;
    float s = 0.f;
#pragma unroll 8
    for (int d = 0; d < DK; ++d) s += q[d] * rk[d];
    QR[gid] = s;
}

// ---------------------------------------------------------------------------
// Fused attention. Block = (b, 32 query rows, half of the heads). Loops its 8
// heads so the head-mean partial is accumulated with plain per-thread global
// RMW into a private buffer (no atomics -> deterministic). Scores in LDS.
// V tiles are streamed by the Tensor Data Mover, double-buffered.
__global__ __launch_bounds__(256)
void k_attn(const __bf16* __restrict__ Qb, const __bf16* __restrict__ Kb,
            const __bf16* __restrict__ Vb, const float* __restrict__ QR,
            const float* __restrict__ relv,
            __bf16* __restrict__ ctx, float* __restrict__ meanPart) {
    __shared__ float sc[QT][S_LEN];                    // 128 KB scores/probs
    __shared__ __align__(32) __bf16 kst[64][64];       // 8 KB  K^T tile [d][j]
    __shared__ float wbuf[QT][72];                     // 9 KB  rel_v weights
#if USE_TDM
    __shared__ __align__(32) __bf16 vst[2][32][64];    // 8 KB  V tiles (TDM)
#endif

    const int tid = threadIdx.x;
    const int w = tid >> 5, lane = tid & 31;
    const int lhalf = lane >> 4, lm = lane & 15;
    const int b  = blockIdx.x / (S_LEN / QT);
    const int i0 = (blockIdx.x % (S_LEN / QT)) * QT;
    const int h0 = blockIdx.y * HPB;

    const int mt = w & 1;    // M subtile of this wave
    const int nt = w >> 1;   // N subtile of this wave

    float* meanBase = meanPart + (size_t)blockIdx.y * ((size_t)B_SZ * S_LEN * S_LEN);

    for (int hh = 0; hh < HPB; ++hh) {
        const int h = h0 + hh;
        // ---- Q fragments for this wave's 16 rows (reused across all keys)
        const __bf16* qrow =
            Qb + ((size_t)(b * S_LEN + i0 + mt * 16 + lm)) * D_MOD + h * DK;
        v16bf qa0 = load_a_frag(qrow,      lhalf);   // dk 0..31
        v16bf qa1 = load_a_frag(qrow + 32, lhalf);   // dk 32..63

        // ---- scores = Q K^T / sqrt(dk), 64 keys staged (transposed) per iter
        for (int kb = 0; kb < S_LEN / 64; ++kb) {
            __syncthreads();
            {   // stage K^T: kst[d][j] = Kb[b, kb*64+j, h*64+d]
                int j = tid >> 2;
                int dbase = (tid & 3) * 16;
                v16bf kv = *reinterpret_cast<const v16bf*>(
                    Kb + ((size_t)(b * S_LEN + kb * 64 + j)) * D_MOD + h * DK + dbase);
#pragma unroll
                for (int e = 0; e < 16; ++e) kst[dbase + e][j] = kv[e];
            }
            __syncthreads();
            v8f acc = {};
            v16bf b0, b1;
#pragma unroll
            for (int e = 0; e < 16; ++e) { int kk = (e < 8) ? (lhalf * 8 + e) : (16 + lhalf * 8 + (e - 8)); b0[e] = kst[kk][nt * 16 + lm]; b1[e] = kst[32 + kk][nt * 16 + lm]; }
            acc = wmma_bf16(qa0, b0, acc);
            acc = wmma_bf16(qa1, b1, acc);
#pragma unroll
            for (int v = 0; v < 8; ++v)
                sc[mt * 16 + v + 8 * lhalf][kb * 64 + nt * 16 + lm] = acc[v] * 0.125f;
        }
        __syncthreads();

        // ---- rel_k bias + softmax (+ mean accumulation + rel_v weights)
        {
            const int r = tid >> 3, g = tid & 7;   // 8 threads per row
            const int i = i0 + r;
            const float* qr = QR + ((size_t)((b * H_NUM + h) * S_LEN + i)) * NREL;
            float mx = -1e30f;
            for (int j = g; j < S_LEN; j += 8) {
                int dd = j - i; dd = dd < -32 ? -32 : (dd > 32 ? 32 : dd);
                float sv = sc[r][j] + qr[dd + 32];
                sc[r][j] = sv;
                mx = fmaxf(mx, sv);
            }
            mx = fmaxf(mx, __shfl_xor(mx, 1, 8));
            mx = fmaxf(mx, __shfl_xor(mx, 2, 8));
            mx = fmaxf(mx, __shfl_xor(mx, 4, 8));
            float sum = 0.f, elo = 0.f, ehi = 0.f;
            for (int j = g; j < S_LEN; j += 8) {
                float p = __expf(sc[r][j] - mx);
                sc[r][j] = p;
                sum += p;
                if (j <= i - 32) elo += p;   // clip bucket r=0
                if (j >= i + 32) ehi += p;   // clip bucket r=64
            }
            sum += __shfl_xor(sum, 1, 8); sum += __shfl_xor(sum, 2, 8); sum += __shfl_xor(sum, 4, 8);
            elo += __shfl_xor(elo, 1, 8); elo += __shfl_xor(elo, 2, 8); elo += __shfl_xor(elo, 4, 8);
            ehi += __shfl_xor(ehi, 1, 8); ehi += __shfl_xor(ehi, 2, 8); ehi += __shfl_xor(ehi, 4, 8);
            const float inv = 1.0f / sum;
            float* mrow = meanBase + ((size_t)(b * S_LEN + i)) * S_LEN;
            for (int j = g; j < S_LEN; j += 8) {
                float p = sc[r][j] * inv;
                sc[r][j] = p;
                mrow[j] += p * (1.0f / (float)H_NUM);   // same thread+addr every head
            }
            if (g == 0) { wbuf[r][0] = elo * inv; wbuf[r][64] = ehi * inv; }
            for (int rr = 1 + g; rr < 64; rr += 8) {
                int j = i + rr - 32;
                wbuf[r][rr] = (j >= 0 && j < S_LEN) ? sc[r][j] : 0.f;
            }
        }
        __syncthreads();

        // ---- context = P @ V + rel_v(wbuf)
        {
            const int d = nt * 16 + lm;
            v8f acc = {};
#if USE_TDM
            // TDM-streamed, double-buffered V tiles: 32 keys x 64 dims bf16.
            if (w == 0) {
                tdm_load_tile_bf16(Vb + ((size_t)(b * S_LEN)) * D_MOD + h * DK,
                                   (unsigned)(uintptr_t)&vst[0][0][0],
                                   64, 32, D_MOD);
            }
            for (int kc = 0; kc < S_LEN / 32; ++kc) {
                if (w == 0) {
                    if (kc + 1 < S_LEN / 32) {
                        tdm_load_tile_bf16(
                            Vb + ((size_t)(b * S_LEN + (kc + 1) * 32)) * D_MOD + h * DK,
                            (unsigned)(uintptr_t)&vst[(kc + 1) & 1][0][0],
                            64, 32, D_MOD);
                        __builtin_amdgcn_s_wait_tensorcnt(1);   // tile kc done
                    } else {
                        __builtin_amdgcn_s_wait_tensorcnt(0);
                    }
                }
                __syncthreads();
                v16bf pa;
#pragma unroll
                for (int e = 0; e < 16; ++e) {
                    int kk = (e < 8) ? (lhalf * 8 + e) : (16 + lhalf * 8 + (e - 8));
                    pa[e] = (__bf16)sc[mt * 16 + lm][kc * 32 + kk];
                }
                v16bf vb;
#pragma unroll
                for (int e = 0; e < 16; ++e) { int kk = (e < 8) ? (lhalf * 8 + e) : (16 + lhalf * 8 + (e - 8)); vb[e] = vst[kc & 1][kk][nt * 16 + lm]; }
                acc = wmma_bf16(pa, vb, acc);
                __syncthreads();
            }
#else
            for (int kc = 0; kc < S_LEN / 32; ++kc) {
                v16bf pa;
#pragma unroll
                for (int e = 0; e < 16; ++e) {
                    int kk = (e < 8) ? (lhalf * 8 + e) : (16 + lhalf * 8 + (e - 8));
                    pa[e] = (__bf16)sc[mt * 16 + lm][kc * 32 + kk];
                }
                v16bf vb;
#pragma unroll
                for (int e = 0; e < 16; ++e) { int kk = (e < 8) ? (lhalf * 8 + e) : (16 + lhalf * 8 + (e - 8)); vb[e] = Vb[((size_t)(b * S_LEN + kc * 32 + kk)) * D_MOD + h * DK + nt * 16 + lm]; }
                acc = wmma_bf16(pa, vb, acc);
            }
#endif
            for (int rr = 0; rr < NREL; ++rr) {
                float rv = relv[rr * DK + d];
#pragma unroll
                for (int v = 0; v < 8; ++v)
                    acc[v] += wbuf[mt * 16 + v + 8 * lhalf][rr] * rv;
            }
#pragma unroll
            for (int v = 0; v < 8; ++v) {
                int row = i0 + mt * 16 + v + 8 * lhalf;
                ctx[((size_t)(b * S_LEN + row)) * D_MOD + h * DK + d] = (__bf16)acc[v];
            }
        }
        __syncthreads();
    }
}

// ---------------------------------------------------------------------------
extern "C" void kernel_launch(void* const* d_in, const int* in_sizes, int n_in,
                              void* d_out, int out_size, void* d_ws, size_t ws_size,
                              hipStream_t stream) {
    (void)in_sizes; (void)n_in; (void)out_size; (void)ws_size;

    const float* q_in = (const float*)d_in[0];
    const float* k_in = (const float*)d_in[1];
    const float* v_in = (const float*)d_in[2];
    const float* Wq   = (const float*)d_in[3];
    const float* Wk   = (const float*)d_in[4];
    const float* Wv   = (const float*)d_in[5];
    const float* Wo   = (const float*)d_in[6];
    const float* bo   = (const float*)d_in[7];
    const float* relk = (const float*)d_in[8];
    const float* relv = (const float*)d_in[9];

    const size_t BS  = (size_t)B_SZ * S_LEN;      // 4096
    const size_t BSD = BS * D_MOD;                // 4,194,304
    const size_t BSS = (size_t)B_SZ * S_LEN * S_LEN;
    const size_t DD  = (size_t)D_MOD * D_MOD;
    const size_t QRN = (size_t)B_SZ * H_NUM * S_LEN * NREL;

    char* wsb = (char*)d_ws;
    size_t off = 0;
    auto take = [&](size_t bytes) { char* p = wsb + off; off += (bytes + 255) & ~(size_t)255; return p; };
    __bf16* qbf   = (__bf16*)take(BSD * 2);
    __bf16* kbf   = (__bf16*)take(BSD * 2);
    __bf16* vbf   = (__bf16*)take(BSD * 2);
    __bf16* wqb   = (__bf16*)take(DD * 2);
    __bf16* wkb   = (__bf16*)take(DD * 2);
    __bf16* wvb   = (__bf16*)take(DD * 2);
    __bf16* wob   = (__bf16*)take(DD * 2);
    float*  Qf    = (float*) take(BSD * 4);
    __bf16* Qbf   = (__bf16*)take(BSD * 2);
    __bf16* Kbf   = (__bf16*)take(BSD * 2);
    __bf16* Vbf   = (__bf16*)take(BSD * 2);
    __bf16* ctxbf = (__bf16*)take(BSD * 2);
    float*  QR    = (float*) take(QRN * 4);
    float*  meanP = (float*) take(BSS * 2 * 4);   // two head-group partials

    float* outMain = (float*)d_out;               // [B,S,D]
    float* outMean = (float*)d_out + BSD;         // [B,S,S]

    const int T = 256;
    auto blks = [](size_t n, int t) { return (unsigned)((n + t - 1) / t); };

    // 1) casts
    k_cast_bf16<<<blks(BSD, T), T, 0, stream>>>(q_in, qbf, BSD);
    k_cast_bf16<<<blks(BSD, T), T, 0, stream>>>(k_in, kbf, BSD);
    k_cast_bf16<<<blks(BSD, T), T, 0, stream>>>(v_in, vbf, BSD);
    k_cast_bf16<<<blks(DD, T), T, 0, stream>>>(Wq, wqb, DD);
    k_cast_bf16<<<blks(DD, T), T, 0, stream>>>(Wk, wkb, DD);
    k_cast_bf16<<<blks(DD, T), T, 0, stream>>>(Wv, wvb, DD);
    k_cast_bf16<<<blks(DD, T), T, 0, stream>>>(Wo, wob, DD);

    // 2) projections (M=4096, N=K=1024)
    dim3 ggrid(BS / 64, D_MOD / 64);
    k_gemm_bf16<<<ggrid, T, 0, stream>>>(qbf, wqb, nullptr, Qf, Qbf, (int)BS, D_MOD, D_MOD);
    k_gemm_bf16<<<ggrid, T, 0, stream>>>(kbf, wkb, nullptr, nullptr, Kbf, (int)BS, D_MOD, D_MOD);
    k_gemm_bf16<<<ggrid, T, 0, stream>>>(vbf, wvb, nullptr, nullptr, Vbf, (int)BS, D_MOD, D_MOD);

    // 3) relative-key dot table
    k_qrel<<<blks(QRN, T), T, 0, stream>>>(Qf, relk, QR);

    // 4) zero the head-mean partial buffers (workspace is poisoned)
    k_zero_f32<<<blks(BSS * 2, T), T, 0, stream>>>(meanP, BSS * 2);

    // 5) fused attention: grid = (b,query-tile) x (head half)
    dim3 agrid(B_SZ * (S_LEN / QT), 2);
    k_attn<<<agrid, T, 0, stream>>>(Qbf, Kbf, Vbf, QR, relv, ctxbf, meanP);

    // 6) combine head-mean partials -> d_out mean region
    k_mean_combine<<<blks(BSS, T), T, 0, stream>>>(meanP, meanP + BSS, outMean, BSS);

    // 7) output projection with bias -> d_out
    k_gemm_bf16<<<ggrid, T, 0, stream>>>(ctxbf, wob, bo, outMain, nullptr, (int)BS, D_MOD, D_MOD);
}
